// KnotAttention_85572928405544
// MI455X (gfx1250) — matrix operand order repaired and checked by reading, not compiled, not run on hardware
//
#include <hip/hip_runtime.h>
#include <hip/hip_bf16.h>
#include <math.h>

// ---------------------------------------------------------------------------
// KnotAttention for MI455X (gfx1250, wave32).
//  - Dense projections: v_wmma_f32_16x16x32_f16, 16x64 C tile per wave,
//    64-deep K slabs staged f32 -> LDS by the Tensor Data Mover,
//    double-buffered with s_wait_tensorcnt (8 WMMAs per slab).
//  - Pairwise silu/tanh core: VALU/trans with v_rcp_f32-based silu/tanh.
// ---------------------------------------------------------------------------

typedef __attribute__((ext_vector_type(16))) _Float16 v16h;
typedef __attribute__((ext_vector_type(8)))  float    v8f;
typedef unsigned int u32x4 __attribute__((ext_vector_type(4)));
typedef int          i32x4 __attribute__((ext_vector_type(4)));
typedef int          i32x8 __attribute__((ext_vector_type(8)));

#define FLAG_SILU  1
#define FLAG_ACC   2
#define FLAG_TRANS 4

#define T_  256
#define D_  512
#define H_  8
#define HD_ 64
#define M_  64
#define F_  256   // HID

__device__ __forceinline__ float fast_rcp(float x) { return __builtin_amdgcn_rcpf(x); }
__device__ __forceinline__ float silu_f(float x) {
    // x * sigmoid(x): v_exp + v_rcp (no IEEE divide chain)
    return x * fast_rcp(1.0f + __expf(-x));
}
__device__ __forceinline__ float tanh_half_f(float z) {
    // tanh(z/2) = (1 - e^-z) / (1 + e^-z)
    float e = __expf(-z);
    return (1.0f - e) * fast_rcp(1.0f + e);
}

// ---------------------------------------------------------------------------
// Tensor Data Mover: 2-D tile (rows x cols of f32) global -> LDS.
// D# group0/group1 per CDNA5 ISA 8.3/8.4; groups 2/3 zero (2-D tensor).
// ---------------------------------------------------------------------------
__device__ __forceinline__ void tdm_load_2d_f32(unsigned int lds_off,
                                                const float* gptr,
                                                int rows, int cols,
                                                int stride_elems)
{
    unsigned long long ga = (unsigned long long)(__SIZE_TYPE__)gptr;
    u32x4 g0;
    g0[0] = 1u;                                            // count=1, user mode
    g0[1] = lds_off;                                       // lds_addr (bytes)
    g0[2] = (unsigned int)ga;                              // global_addr[31:0]
    g0[3] = (unsigned int)((ga >> 32) & 0x1FFFFFFu)        // global_addr[56:32]
          | (2u << 30);                                    // type = 2 (image)

    unsigned int d0 = (unsigned int)cols;                  // tensor_dim0
    unsigned int d1 = (unsigned int)rows;                  // tensor_dim1
    unsigned long long s0 = (unsigned long long)stride_elems;
    i32x8 g1;
    g1[0] = (int)(2u << 16);                               // data_size = 2 (4B)
    g1[1] = (int)((d0 & 0xFFFFu) << 16);                   // dim0[15:0] @ bits 63:48
    g1[2] = (int)((d0 >> 16) | ((d1 & 0xFFFFu) << 16));    // dim0[31:16], dim1[15:0]
    g1[3] = (int)((d1 >> 16) | (((unsigned)cols & 0xFFFFu) << 16)); // dim1 hi, tile_dim0
    g1[4] = (int)((unsigned)rows & 0xFFFFu);               // tile_dim1 (tile_dim2=0)
    g1[5] = (int)(unsigned int)(s0 & 0xFFFFFFFFu);         // dim0_stride[31:0]
    g1[6] = (int)(unsigned int)((s0 >> 32) & 0xFFFFu);     // dim0_stride[47:32]
    g1[7] = 0;

    i32x4 z4 = {0, 0, 0, 0};
#if __clang_major__ >= 23
    i32x8 z8 = {0, 0, 0, 0, 0, 0, 0, 0};
    __builtin_amdgcn_tensor_load_to_lds(g0, g1, z4, z4, z8, 0);
#else
    __builtin_amdgcn_tensor_load_to_lds(g0, g1, z4, z4, 0);
#endif
}

// ---------------------------------------------------------------------------
// f32 GEMM via f16 WMMA. One wave (32 threads) computes a 16x64 C tile;
// K consumed in 64-deep slabs (2 x 16x16x32 per N-subtile = 8 WMMAs/slab).
// Slabs DMA'd to LDS by TDM, double buffered.
// grid = (N/64, M/16, batch); M mult 16, N mult 64, K mult 64.
// flags: bit0 silu(out), bit1 accumulate into C, bit2 store C transposed.
// ---------------------------------------------------------------------------
__global__ void wmma_gemm_kernel(const float* __restrict__ A, int lda, long long strideA,
                                 const float* __restrict__ B, int ldb, long long strideB,
                                 const float* __restrict__ bias,
                                 float* __restrict__ C, int ldc, long long strideC,
                                 int K, int flags)
{
    __shared__ float Asf[2][16 * 64];   // [buf][row][k]   4 KB each
    __shared__ float Bsf[2][64 * 64];   // [buf][k][col]  16 KB each

    const int lane  = threadIdx.x;      // one wave32
    const int tileN = blockIdx.x * 64;
    const int tileM = blockIdx.y * 16;
    const int bidx  = blockIdx.z;

    A += (long long)bidx * strideA;
    B += (long long)bidx * strideB;
    C += (long long)bidx * strideC;

    const int hi   = lane >> 4;
    const int lrow = lane & 15;

    v8f acc0 = {}, acc1 = {}, acc2 = {}, acc3 = {};

    const int ktiles = K >> 6;          // 64-deep slabs
    const float* Abase = A + (long long)tileM * lda;

    // prologue: DMA first K-slab into buffer 0
    tdm_load_2d_f32((unsigned int)(__SIZE_TYPE__)&Asf[0][0], Abase, 16, 64, lda);
    tdm_load_2d_f32((unsigned int)(__SIZE_TYPE__)&Bsf[0][0], B + tileN, 64, 64, ldb);

    for (int kt = 0; kt < ktiles; ++kt) {
        const int buf = kt & 1;
        if (kt + 1 < ktiles) {
            const int k1 = (kt + 1) << 6;
            tdm_load_2d_f32((unsigned int)(__SIZE_TYPE__)&Asf[buf ^ 1][0],
                            Abase + k1, 16, 64, lda);
            tdm_load_2d_f32((unsigned int)(__SIZE_TYPE__)&Bsf[buf ^ 1][0],
                            B + (long long)k1 * ldb + tileN, 64, 64, ldb);
            __builtin_amdgcn_s_wait_tensorcnt((short)2);   // current slab ready
        } else {
            __builtin_amdgcn_s_wait_tensorcnt((short)0);
        }

        #pragma unroll
        for (int ks = 0; ks < 64; ks += 32) {
            // ---- A fragment: K contiguous in groups of 8 -> float4 LDS loads
            const float* ar = &Asf[buf][lrow * 64 + ks + hi * 8];
            float4 a0 = *(const float4*)(ar + 0);
            float4 a1 = *(const float4*)(ar + 4);
            float4 a2 = *(const float4*)(ar + 16);
            float4 a3 = *(const float4*)(ar + 20);
            v16h af;
            af[0]=(_Float16)a0.x; af[1]=(_Float16)a0.y; af[2]=(_Float16)a0.z; af[3]=(_Float16)a0.w;
            af[4]=(_Float16)a1.x; af[5]=(_Float16)a1.y; af[6]=(_Float16)a1.z; af[7]=(_Float16)a1.w;
            af[8]=(_Float16)a2.x; af[9]=(_Float16)a2.y; af[10]=(_Float16)a2.z; af[11]=(_Float16)a2.w;
            af[12]=(_Float16)a3.x; af[13]=(_Float16)a3.y; af[14]=(_Float16)a3.z; af[15]=(_Float16)a3.w;

            // ---- build all 4 B fragments into distinct registers first
            //      (avoids WMMA->VALU WAR hazard nops between the WMMAs)
            v16h bf0, bf1, bf2, bf3;
            #pragma unroll
            for (int e = 0; e < 16; ++e) {
                const int krow = (ks + (e & 7) + hi * 8 + (e >> 3) * 16) * 64 + lrow;
                bf0[e] = (_Float16)Bsf[buf][krow + 0];
                bf1[e] = (_Float16)Bsf[buf][krow + 16];
                bf2[e] = (_Float16)Bsf[buf][krow + 32];
                bf3[e] = (_Float16)Bsf[buf][krow + 48];
            }
            acc0 = __builtin_amdgcn_wmma_f32_16x16x32_f16(false, af, false, bf0, (short)0, acc0, false, false);
            acc1 = __builtin_amdgcn_wmma_f32_16x16x32_f16(false, af, false, bf1, (short)0, acc1, false, false);
            acc2 = __builtin_amdgcn_wmma_f32_16x16x32_f16(false, af, false, bf2, (short)0, acc2, false, false);
            acc3 = __builtin_amdgcn_wmma_f32_16x16x32_f16(false, af, false, bf3, (short)0, acc3, false, false);
        }
    }

    // ---- epilogue: C/D layout = 8 rows per lane, halves split at lane 16 ----
    #pragma unroll
    for (int tn = 0; tn < 4; ++tn) {
        const v8f acc = (tn == 0) ? acc0 : (tn == 1) ? acc1 : (tn == 2) ? acc2 : acc3;
        const int colg = tileN + tn * 16 + lrow;
        const float bval = bias ? bias[colg] : 0.0f;
        #pragma unroll
        for (int e = 0; e < 8; ++e) {
            int   row = tileM + e + hi * 8;
            float v   = acc[e] + bval;
            if (flags & FLAG_SILU) v = silu_f(v);
            long long addr = (flags & FLAG_TRANS)
                                 ? ((long long)colg * ldc + row)
                                 : ((long long)row * ldc + colg);
            if (flags & FLAG_ACC) v += C[addr];
            C[addr] = v;
        }
    }
}

// ---------------------------------------------------------------------------
// Block reductions (256 threads).
// ---------------------------------------------------------------------------
__device__ __forceinline__ float blockReduceSum256(float v, float* s) {
    int t = threadIdx.x;
    s[t] = v; __syncthreads();
    #pragma unroll
    for (int k = 128; k > 0; k >>= 1) {
        if (t < k) s[t] += s[t + k];
        __syncthreads();
    }
    float r = s[0]; __syncthreads();
    return r;
}
__device__ __forceinline__ float blockReduceMax256(float v, float* s) {
    int t = threadIdx.x;
    s[t] = v; __syncthreads();
    #pragma unroll
    for (int k = 128; k > 0; k >>= 1) {
        if (t < k) s[t] = fmaxf(s[t], s[t + k]);
        __syncthreads();
    }
    float r = s[0]; __syncthreads();
    return r;
}

// ---------------------------------------------------------------------------
// Pairwise silu MLP + linking + masked softmax + attn@v.
// grid = (T, H), block = 256.  a_i row staged to LDS by TDM (wave 0).
// aJt layout [h][f][t] -> coalesced across j-threads.
// ---------------------------------------------------------------------------
__global__ void attn_pairwise_kernel(const float* __restrict__ aI,
                                     const float* __restrict__ aJt,
                                     const float* __restrict__ vmat,
                                     const float* __restrict__ W2,
                                     const float* __restrict__ b2,
                                     float* __restrict__ outAttn,
                                     float* __restrict__ wPart,
                                     float* __restrict__ aPart)
{
    __shared__ float sAi[F_];
    __shared__ float sW2[F_];
    __shared__ float sP[T_];
    __shared__ float sRed[T_];

    const int j = threadIdx.x;   // 0..255
    const int i = blockIdx.x;
    const int h = blockIdx.y;

    if (j < 32) {   // wave 0 DMAs the a_i row into LDS
        tdm_load_2d_f32((unsigned int)(__SIZE_TYPE__)&sAi[0],
                        aI + ((long long)h * T_ + i) * F_, 1, F_, F_);
        __builtin_amdgcn_s_wait_tensorcnt((short)0);
    }
    sW2[j] = W2[j * 2 + 0] - W2[j * 2 + 1];   // 2-class softmax collapses to diff
    __syncthreads();

    const float* colp = aJt + (long long)h * F_ * T_ + j;
    float z = b2[0] - b2[1];
    for (int f0 = 0; f0 < F_; f0 += 32) {
        __builtin_prefetch(colp + (long long)(f0 + 32) * T_, 0, 1);
        #pragma unroll
        for (int f = 0; f < 32; ++f) {
            float xv = sAi[f0 + f] + colp[(long long)(f0 + f) * T_];
            z += sW2[f0 + f] * silu_f(xv);
        }
    }
    // crossings[...,0]-crossings[...,1] = tanh((z0-z1)/2)
    float linking = tanh_half_f(z);

    float s1 = blockReduceSum256(linking, sRed);
    float s2 = blockReduceSum256(fabsf(linking), sRed);
    if (j == 0) { wPart[h * T_ + i] = s1; aPart[h * T_ + i] = s2; }

    // causal masked softmax over j (scores = linking / sqrt(64))
    float score = (j <= i) ? (linking * 0.125f) : -1e30f;
    float mx  = blockReduceMax256(score, sRed);
    float ev  = __expf(score - mx);
    float den = blockReduceSum256(ev, sRed);
    sP[j] = ev * fast_rcp(den);
    __syncthreads();

    // out[i, h*64+d] = sum_j p_j * v[j, h*64+d]
    if (j < HD_) {
        float acc2 = 0.0f;
        #pragma unroll 4
        for (int jj = 0; jj < T_; ++jj)
            acc2 += sP[jj] * vmat[(long long)jj * D_ + h * HD_ + j];
        outAttn[(long long)i * D_ + h * HD_ + j] = acc2;
    }
}

// ---------------------------------------------------------------------------
// Final deterministic reduction of writhe / linking_mean into d_out tail.
// ---------------------------------------------------------------------------
__global__ void finalize_kernel(const float* __restrict__ wPart,
                                const float* __restrict__ aPart,
                                float* __restrict__ outTail)
{
    int t = threadIdx.x;
    if (t < H_) {
        float s = 0.0f;
        for (int i = 0; i < T_; ++i) s += wPart[t * T_ + i];
        outTail[t] = s;
    }
    if (t == H_) {
        float s = 0.0f;
        for (int k = 0; k < H_ * T_; ++k) s += aPart[k];
        outTail[H_] = s * (1.0f / ((float)H_ * T_ * T_));
    }
}

// ---------------------------------------------------------------------------
extern "C" void kernel_launch(void* const* d_in, const int* in_sizes, int n_in,
                              void* d_out, int out_size, void* d_ws, size_t ws_size,
                              hipStream_t stream)
{
    const float* x      = (const float*)d_in[0];
    const float* coords = (const float*)d_in[1];
    const float* Wq     = (const float*)d_in[2];
    const float* Wk     = (const float*)d_in[3];
    const float* Wv     = (const float*)d_in[4];
    const float* Wo     = (const float*)d_in[5];
    const float* W1     = (const float*)d_in[6];
    const float* b1     = (const float*)d_in[7];
    const float* W2     = (const float*)d_in[8];
    const float* b2     = (const float*)d_in[9];
    const float* W3     = (const float*)d_in[10];
    const float* b3     = (const float*)d_in[11];
    const float* W4     = (const float*)d_in[12];
    const float* b4     = (const float*)d_in[13];
    float* out = (float*)d_out;

    float* ws    = (float*)d_ws;
    float* q     = ws;  ws += T_ * D_;
    float* k     = ws;  ws += T_ * D_;
    float* v     = ws;  ws += T_ * D_;
    float* aI    = ws;  ws += H_ * T_ * F_;   // [h][t][f]
    float* aJt   = ws;  ws += H_ * T_ * F_;   // [h][f][t]
    float* attnO = ws;  ws += T_ * D_;
    float* tmp1  = ws;  ws += T_ * D_;
    float* tmp2  = ws;  ws += T_ * D_;
    float* wPart = ws;  ws += H_ * T_;
    float* aPart = ws;  ws += H_ * T_;

    const dim3 blk(32);
    const long long sHead = (long long)T_ * F_;

    // q/k/v = x @ W{q,k,v}   (M=256, N=512, K=512)
    dim3 gQKV(D_ / 64, T_ / 16, 1);
    wmma_gemm_kernel<<<gQKV, blk, 0, stream>>>(x, D_, 0, Wq, D_, 0, nullptr, q, D_, 0, D_, 0);
    wmma_gemm_kernel<<<gQKV, blk, 0, stream>>>(x, D_, 0, Wk, D_, 0, nullptr, k, D_, 0, D_, 0);
    wmma_gemm_kernel<<<gQKV, blk, 0, stream>>>(x, D_, 0, Wv, D_, 0, nullptr, v, D_, 0, D_, 0);

    // per-head projections (batched over blockIdx.z = head, A column offset 64/head)
    dim3 gHP(F_ / 64, T_ / 16, H_);
    // aI = q_h @ W1[0:64]         then  aI += coords @ W1[128:192]
    wmma_gemm_kernel<<<gHP, blk, 0, stream>>>(q,      D_, 64, W1,            F_, 0, nullptr, aI,  F_, sHead, HD_, 0);
    wmma_gemm_kernel<<<gHP, blk, 0, stream>>>(coords, M_, 0,  W1 + 128 * F_, F_, 0, nullptr, aI,  F_, sHead, M_,  FLAG_ACC);
    // aJ^T = (k_h @ W1[64:128])^T then  aJ^T += (coords @ W1[192:256] + b1)^T
    wmma_gemm_kernel<<<gHP, blk, 0, stream>>>(k,      D_, 64, W1 + 64 * F_,  F_, 0, nullptr, aJt, T_, sHead, HD_, FLAG_TRANS);
    wmma_gemm_kernel<<<gHP, blk, 0, stream>>>(coords, M_, 0,  W1 + 192 * F_, F_, 0, b1,      aJt, T_, sHead, M_,  FLAG_ACC | FLAG_TRANS);

    // pairwise MLP + softmax + attn@v
    attn_pairwise_kernel<<<dim3(T_, H_), dim3(256), 0, stream>>>(
        aI, aJt, v, W2, b2, attnO, wPart, aPart);

    // output MLP: silu(attnO@W3+b3) @ W4 + b4, then @ Wo -> d_out
    wmma_gemm_kernel<<<gQKV, blk, 0, stream>>>(attnO, D_, 0, W3, D_, 0, b3,      tmp1, D_, 0, D_, FLAG_SILU);
    wmma_gemm_kernel<<<gQKV, blk, 0, stream>>>(tmp1,  D_, 0, W4, D_, 0, b4,      tmp2, D_, 0, D_, 0);
    wmma_gemm_kernel<<<gQKV, blk, 0, stream>>>(tmp2,  D_, 0, Wo, D_, 0, nullptr, out,  D_, 0, D_, 0);

    // writhe[8] + linking_mean -> d_out tail
    finalize_kernel<<<1, 32, 0, stream>>>(wPart, aPart, out + T_ * D_);
}
